// PriorIntegrationModule_58428735095585
// MI455X (gfx1250) — matrix operand order repaired and checked by reading, not compiled, not run on hardware
//
#include <hip/hip_runtime.h>

typedef _Float16 v16h __attribute__((ext_vector_type(16)));
typedef _Float16 v8h  __attribute__((ext_vector_type(8)));
typedef float    v8f  __attribute__((ext_vector_type(8)));

#define BATCH 4
#define CCH   512
#define NPIX  4096
#define CQ    64

// padded LDS row strides (halves) -> conflict-free 16-lane b128 reads, 16B aligned
#define VST 40   // V tile rows (32 data halves + 8 pad)
#define KST 72   // K tile rows (64 data halves + 8 pad)
#define PST 40   // P tile rows (32 data halves + 8 pad)

static __device__ __forceinline__ v8f wmma32f16(v16h a, v16h b, v8f c) {
  return __builtin_amdgcn_wmma_f32_16x16x32_f16(false, a, false, b, (short)0, c, false, false);
}
static __device__ __forceinline__ v16h cat16(v8h lo, v8h hi) {
  return __builtin_shufflevector(lo, hi, 0,1,2,3,4,5,6,7,8,9,10,11,12,13,14,15);
}

// ---------------- projection kernels (fp32 math -> f16 fragments) ----------------

// qh[b][m][d] = Wq(64x512) @ x[b][:, m] + bq      (A-matrix layout: row-major, d contiguous)
__global__ __launch_bounds__(256) void qproj(const float* __restrict__ x,
                                             const float* __restrict__ Wq,
                                             const float* __restrict__ bq,
                                             _Float16* __restrict__ qh) {
  __shared__ float w[64 * 65];
  int t = threadIdx.x;
  int o = t & 63, nl = t >> 6;                 // 64 outputs x 4 pixels per block
  int bn = blockIdx.x * 4 + nl;                // b*N + n
  int b = bn >> 12, n = bn & (NPIX - 1);
  const float* xb = x + ((size_t)b * CCH) * NPIX + n;
  float acc = 0.f;
  for (int cc = 0; cc < CCH; cc += 64) {
    __syncthreads();
#pragma unroll
    for (int i = 0; i < 16; i++) {             // stage Wq[:, cc:cc+64] in LDS (padded)
      int e = t * 16 + i;
      int ro = e >> 6, j = e & 63;
      w[ro * 65 + j] = Wq[ro * CCH + cc + j];
    }
    __syncthreads();
#pragma unroll 8
    for (int j = 0; j < 64; j++)
      acc += w[o * 65 + j] * xb[(size_t)(cc + j) * NPIX];
  }
  acc += bq[o];
  qh[((size_t)bn << 6) + o] = (_Float16)acc;
}

// kh[b][n][d] = Wk(64x64) @ prior[b][:, n] + bk   (B-matrix layout: d contiguous per column)
__global__ __launch_bounds__(256) void kproj(const float* __restrict__ prior,
                                             const float* __restrict__ Wk,
                                             const float* __restrict__ bk,
                                             _Float16* __restrict__ kh) {
  __shared__ float w[64 * 65];
  int t = threadIdx.x;
  int d = t & 63, nl = t >> 6;
  int bn = blockIdx.x * 4 + nl;
  int b = bn >> 12, n = bn & (NPIX - 1);
#pragma unroll
  for (int i = 0; i < 16; i++) {
    int e = t * 16 + i;
    int ro = e >> 6, j = e & 63;
    w[ro * 65 + j] = Wk[ro * CQ + j];
  }
  __syncthreads();
  const float* pb = prior + ((size_t)b * CQ) * NPIX + n;
  float acc = 0.f;
#pragma unroll 8
  for (int j = 0; j < 64; j++)
    acc += w[d * 65 + j] * pb[(size_t)j * NPIX];
  acc += bk[d];
  kh[((size_t)bn << 6) + d] = (_Float16)acc;
}

// vh[b][c][n] = Wv(512x64) @ prior[b][:, n] + bv  (A-matrix layout for M=c, K=n: n contiguous)
__global__ __launch_bounds__(256) void vproj(const float* __restrict__ prior,
                                             const float* __restrict__ Wv,
                                             const float* __restrict__ bv,
                                             _Float16* __restrict__ vh) {
  int bc = blockIdx.x;                          // b*512 + c
  int b = bc >> 9, c = bc & (CCH - 1);
  int n = blockIdx.y * 256 + threadIdx.x;
  const float* pb = prior + ((size_t)b * CQ) * NPIX + n;
  const float* wv = Wv + c * CQ;
  float acc = 0.f;
#pragma unroll 8
  for (int j = 0; j < 64; j++)
    acc += wv[j] * pb[(size_t)j * NPIX];
  acc += bv[c];
  vh[((size_t)bc << 12) + n] = (_Float16)acc;
}

// ---------------- pass 1: softmax row stats (flash, online) ----------------

__global__ __launch_bounds__(128) void attn_stats(const _Float16* __restrict__ qh,
                                                  const _Float16* __restrict__ kh,
                                                  float* __restrict__ rmax,
                                                  float* __restrict__ rsum) {
  int lane = threadIdx.x & 31;
  int wid  = threadIdx.x >> 5;
  int mt = blockIdx.x * 4 + wid;               // global 16-row tile id
  int b = mt >> 8;
  int m0 = (mt & 255) << 4;
  int lrow = lane & 15, lhi = lane >> 4;
  int koffA = lhi ? 16 : 0;                    // A frag K split
  int koffB = lhi ? 16 : 0;                    // B frag K split

  const _Float16* qb = qh + ((size_t)(b * NPIX + m0 + lrow) << 6);
  v16h aq0 = cat16(*(const v8h*)(qb + koffA),      *(const v8h*)(qb + 8 + koffA));
  v16h aq1 = cat16(*(const v8h*)(qb + 32 + koffA), *(const v8h*)(qb + 40 + koffA));

  float mx[8], sm[8];
#pragma unroll
  for (int r = 0; r < 8; r++) { mx[r] = -1e30f; sm[r] = 0.f; }

  const _Float16* kb = kh + ((size_t)(b * NPIX) << 6);
  for (int n0 = 0; n0 < NPIX; n0 += 16) {
    const _Float16* kp = kb + ((size_t)(n0 + lrow) << 6) + koffB;
    v16h bk0 = cat16(*(const v8h*)(kp),      *(const v8h*)(kp + 8));
    v16h bk1 = cat16(*(const v8h*)(kp + 32), *(const v8h*)(kp + 40));
    v8f e = {};
    e = wmma32f16(aq0, bk0, e);
    e = wmma32f16(aq1, bk1, e);
#pragma unroll
    for (int r = 0; r < 8; r++) {
      float v = e[r];
      float nm = fmaxf(mx[r], v);
      sm[r] = sm[r] * __expf(mx[r] - nm) + __expf(v - nm);
      mx[r] = nm;
    }
  }
  // merge the 16 lanes of each half (each lane held a subset of columns)
#pragma unroll
  for (int mask = 1; mask < 16; mask <<= 1) {
#pragma unroll
    for (int r = 0; r < 8; r++) {
      float omx = __shfl_xor(mx[r], mask, 32);
      float osm = __shfl_xor(sm[r], mask, 32);
      float nm = fmaxf(mx[r], omx);
      sm[r] = sm[r] * __expf(mx[r] - nm) + osm * __expf(omx - nm);
      mx[r] = nm;
    }
  }
  if (lrow == 0) {
#pragma unroll
    for (int r = 0; r < 8; r++) {
      int m = m0 + r + 8 * lhi;
      rmax[b * NPIX + m] = mx[r];
      rsum[b * NPIX + m] = sm[r];
    }
  }
}

// ---------------- pass 2: out = V @ P^T with block-shared LDS V/K tiles ----------------
// Block: 256 threads = 8 waves. All waves share (b, c-half); wave w owns m-tile m0+16w.
// Per 32-n step: stage V(256c x 32n) and K(32n x 64d) once in LDS (double buffered,
// global->reg loads overlap compute), every wave reads fragments from LDS.

__global__ __launch_bounds__(256) void attn_out(const _Float16* __restrict__ qh,
                                                const _Float16* __restrict__ kh,
                                                const _Float16* __restrict__ vh,
                                                const float* __restrict__ rmax,
                                                const float* __restrict__ rsum,
                                                const float* __restrict__ x,
                                                const float* __restrict__ gamma,
                                                float* __restrict__ outp) {
  __shared__ __attribute__((aligned(16))) _Float16 vs[2][256 * VST];  // 2 x 20 KB
  __shared__ __attribute__((aligned(16))) _Float16 ks[2][32 * KST];   // 2 x 4.5 KB
  __shared__ __attribute__((aligned(16))) _Float16 pb[8][16 * PST];   // 8 x 1.25 KB

  int t    = threadIdx.x;
  int lane = t & 31;
  int wid  = t >> 5;
  int b    = blockIdx.y;
  int ch   = blockIdx.x & 1;
  int mblk = blockIdx.x >> 1;
  int m0   = mblk * 128 + wid * 16;
  int c0   = ch * 256;
  int lrow = lane & 15, lhi = lane >> 4;
  int koffA = lhi ? 16 : 0;
  int koffB = lhi ? 16 : 0;
  _Float16* pw = pb[wid];

  const _Float16* qb = qh + ((size_t)(b * NPIX + m0 + lrow) << 6);
  v16h aq0 = cat16(*(const v8h*)(qb + koffA),      *(const v8h*)(qb + 8 + koffA));
  v16h aq1 = cat16(*(const v8h*)(qb + 32 + koffA), *(const v8h*)(qb + 40 + koffA));

  float mxr[8], ism[8];
#pragma unroll
  for (int r = 0; r < 8; r++) {
    int m = m0 + r + 8 * lhi;
    mxr[r] = rmax[b * NPIX + m];
    ism[r] = 1.0f / rsum[b * NPIX + m];
  }

  v8f acc[16];
#pragma unroll
  for (int ct = 0; ct < 16; ct++) acc[ct] = (v8f){};

  // staging addresses: thread t owns V row c_local=t (4x b128) and K chunk (1x b128)
  const _Float16* vsrc0 = vh + ((size_t)(b * CCH + c0 + t) << 12);
  const _Float16* ksrc0 = kh + ((size_t)(b * NPIX + (t >> 3)) << 6) + 8 * (t & 7);

  float4 vr[4], kr;
  // prologue: loads for n0 = 0
#pragma unroll
  for (int i = 0; i < 4; i++) vr[i] = *(const float4*)(vsrc0 + 8 * i);
  kr = *(const float4*)(ksrc0);

  for (int n0 = 0; n0 < NPIX; n0 += 32) {
    int buf = (n0 >> 5) & 1;
    // commit staged regs to LDS
    _Float16* vrow = &vs[buf][t * VST];
#pragma unroll
    for (int i = 0; i < 4; i++) *(float4*)(vrow + 8 * i) = vr[i];
    *(float4*)(&ks[buf][(t >> 3) * KST + 8 * (t & 7)]) = kr;
    __syncthreads();

    // issue next step's global loads (overlap with compute below)
    if (n0 + 32 < NPIX) {
      const _Float16* vsrc = vsrc0 + n0 + 32;
#pragma unroll
      for (int i = 0; i < 4; i++) vr[i] = *(const float4*)(vsrc + 8 * i);
      kr = *(const float4*)(ksrc0 + ((size_t)(n0 + 32) << 6));
    }

    // energy tiles at n-local 0 and 16 (M=m, N=n), K frags from LDS
    v8f e0 = {}, e1 = {};
    {
      const _Float16* kp = &ks[buf][lrow * KST + koffB];
      e0 = wmma32f16(aq0, cat16(*(const v8h*)(kp),      *(const v8h*)(kp + 8)),  e0);
      e0 = wmma32f16(aq1, cat16(*(const v8h*)(kp + 32), *(const v8h*)(kp + 40)), e0);
      const _Float16* kq = &ks[buf][(16 + lrow) * KST + koffB];
      e1 = wmma32f16(aq0, cat16(*(const v8h*)(kq),      *(const v8h*)(kq + 8)),  e1);
      e1 = wmma32f16(aq1, cat16(*(const v8h*)(kq + 32), *(const v8h*)(kq + 40)), e1);
    }
    // P = exp(E - max) / sum  -> wave-private LDS tile [m][k], k local 32
#pragma unroll
    for (int r = 0; r < 8; r++) {
      float p0 = __expf(e0[r] - mxr[r]) * ism[r];
      float p1 = __expf(e1[r] - mxr[r]) * ism[r];
      int row = (r + 8 * lhi) * PST;
      pw[row + lrow]      = (_Float16)p0;
      pw[row + 16 + lrow] = (_Float16)p1;
    }
    // B-fragment of P^T (K = n local 32, N = m)
    const _Float16* pr = pw + lrow * PST + koffB;
    v16h bp = cat16(*(const v8h*)(pr), *(const v8h*)(pr + 8));
    // D(M=c, N=m) += V(c x n) * P^T(n x m); V A-fragments from shared LDS tile
#pragma unroll
    for (int ct = 0; ct < 16; ct++) {
      const _Float16* vp = &vs[buf][(ct * 16 + lrow) * VST];
      v16h av = cat16(*(const v8h*)(vp + koffA), *(const v8h*)(vp + 8 + koffA));
      acc[ct] = wmma32f16(av, bp, acc[ct]);
    }
    __syncthreads();
  }

  float g = gamma[0];
#pragma unroll
  for (int ct = 0; ct < 16; ct++) {
#pragma unroll
    for (int r = 0; r < 8; r++) {
      int c = c0 + ct * 16 + r + 8 * lhi;
      size_t idx = ((size_t)(b * CCH + c) << 12) + m0 + lrow;   // coalesced in m
      outp[idx] = g * acc[ct][r] + x[idx];
    }
  }
}

// ---------------- launch ----------------

extern "C" void kernel_launch(void* const* d_in, const int* in_sizes, int n_in,
                              void* d_out, int out_size, void* d_ws, size_t ws_size,
                              hipStream_t stream) {
  (void)in_sizes; (void)n_in; (void)out_size;
  const float* x     = (const float*)d_in[0];
  const float* prior = (const float*)d_in[1];
  const float* Wq    = (const float*)d_in[2];
  const float* bq    = (const float*)d_in[3];
  const float* Wk    = (const float*)d_in[4];
  const float* bk    = (const float*)d_in[5];
  const float* Wv    = (const float*)d_in[6];
  const float* bv    = (const float*)d_in[7];
  const float* gamma = (const float*)d_in[8];
  float* outp = (float*)d_out;

  size_t qh_e = (size_t)BATCH * NPIX * CQ;        // 1M halves
  size_t vh_e = (size_t)BATCH * CCH * NPIX;       // 8M halves
  _Float16* qh = (_Float16*)d_ws;
  _Float16* kh = qh + qh_e;
  _Float16* vh = kh + qh_e;
  float* rmax = (float*)(vh + vh_e);
  float* rsum = rmax + (size_t)BATCH * NPIX;
  size_t need = (2 * qh_e + vh_e) * sizeof(_Float16) + 2 * (size_t)BATCH * NPIX * sizeof(float);
  if (ws_size < need) return;

  qproj<<<BATCH * NPIX / 4, 256, 0, stream>>>(x, Wq, bq, qh);
  kproj<<<BATCH * NPIX / 4, 256, 0, stream>>>(prior, Wk, bk, kh);
  vproj<<<dim3(BATCH * CCH, NPIX / 256), 256, 0, stream>>>(prior, Wv, bv, vh);
  attn_stats<<<BATCH * NPIX / 16 / 4, 128, 0, stream>>>(qh, kh, rmax, rsum);
  // grid.x = (N/128 m-blocks) * (2 c-halves), grid.y = batch
  attn_out<<<dim3((NPIX / 128) * 2, BATCH), 256, 0, stream>>>(qh, kh, vh, rmax, rsum, x, gamma, outp);
}
